// StackEncoder_10136122818941
// MI455X (gfx1250) — matrix operand order repaired and checked by reading, not compiled, not run on hardware
//
#include <hip/hip_runtime.h>
#include <hip/hip_bf16.h>

#define B_  128
#define L_  64
#define D_  1024
#define K_  2048          // concat(h_l, h_r)
#define NW_ 5120          // 5*D gate columns
#define NSTEP 63

#define KC_     128                   // K chunk per TDM transfer
#define NCHUNK  (K_ / KC_)            // 16
#define ROWS_   80                    // 5 gates x 16 columns per block panel
#define ROWB_   (KC_ * 2)             // 256 B of data per LDS row
#define RSTRIDE (ROWB_ + 16)          // 272 B with TDM pad -> conflict-free banks
#define BUFB_   (ROWS_ * RSTRIDE)     // 21760 B per LDS buffer

typedef __attribute__((ext_vector_type(16))) __bf16 v16bf;
typedef __attribute__((ext_vector_type(8)))  __bf16 v8bf;
typedef __attribute__((ext_vector_type(8)))  float  v8f;
typedef __attribute__((ext_vector_type(4)))  unsigned int u32x4;
typedef __attribute__((ext_vector_type(8)))  int i32x8;
typedef __attribute__((ext_vector_type(4)))  int i32x4;

#if __has_builtin(__builtin_amdgcn_tensor_load_to_lds)
#define HAVE_TDM 1
#else
#define HAVE_TDM 0
#endif

#if __has_builtin(__builtin_amdgcn_s_wait_tensorcnt)
#define WAIT_TENSOR(n) __builtin_amdgcn_s_wait_tensorcnt(n)
#else
#define WAIT_TENSOR(n) asm volatile("s_wait_tensorcnt %0" ::"i"(n) : "memory")
#endif

__device__ __forceinline__ float sigmoidf(float x) {
    return 1.0f / (1.0f + __expf(-x));
}

#if HAVE_TDM
// Issue one TDM DMA: 2D tile (128 K-elements x 80 rows, bf16) global -> LDS.
// Row stride in global = 2048 elements; LDS rows padded +16B per 256B so the
// 16 rows read by a half-wave land in distinct bank groups.
__device__ __forceinline__ void tdm_load_tile(const __bf16* gsrc, unsigned lds_off) {
    const unsigned long long ga = (unsigned long long)gsrc;
    u32x4 g0;
    g0[0] = 1u;                                   // count=1, user descriptor
    g0[1] = lds_off;                              // LDS byte address
    g0[2] = (unsigned)ga;                         // global addr [31:0]
    g0[3] = (unsigned)(ga >> 32) | (2u << 30);    // addr [56:32] | type=2
    i32x8 g1;
    g1[0] = (1 << 16)          // data_size = 2 bytes
          | (1 << 20)          // pad_enable
          | (5 << 22)          // pad_interval: every 64 DWORD (=256B row)
          | (3 << 25);         // pad_amount: 4 DWORD (=16B)
    g1[1] = (int)((K_ & 0xFFFF) << 16);                 // tensor_dim0 lo16
    g1[2] = (int)((K_ >> 16) | (ROWS_ << 16));          // td0 hi16 | td1 lo16
    g1[3] = (int)(KC_ << 16);                           // td1 hi16 | tile_dim0
    g1[4] = (int)ROWS_;                                 // tile_dim1 | tile_dim2=0
    g1[5] = (int)K_;                                    // tensor_dim0_stride lo32
    g1[6] = 0;                                          // stride0 hi | stride1 lo
    g1[7] = 0;
    const i32x4 z4 = {0, 0, 0, 0};
#if defined(__clang_major__) && (__clang_major__ >= 23)
    const i32x8 z8 = {0, 0, 0, 0, 0, 0, 0, 0};
    __builtin_amdgcn_tensor_load_to_lds(g0, g1, z4, z4, z8, 0);
#else
    __builtin_amdgcn_tensor_load_to_lds(g0, g1, z4, z4, 0);
#endif
}
#endif

// ---------------------------------------------------------------------------
// Kernel 1: weight pack. Wpack row r = dt*80 + g*16 + col holds the K=2048
// column (bf16) of gate g, output dim d = dt*16+col, so each block's B panel
// (one dt) is 80 contiguous rows -> a single 2D TDM tile per K chunk.
// ---------------------------------------------------------------------------
__global__ __launch_bounds__(256) void prep_weights(const float* __restrict__ Ul,
                                                    const float* __restrict__ Ur,
                                                    __bf16* __restrict__ Wp) {
    __shared__ float tile[32][33];
    const int k0 = blockIdx.x * 32;   // K_/32 = 64
    const int n0 = blockIdx.y * 32;   // NW_/32 = 160
    const int tx = threadIdx.x & 31;
    const int ty = threadIdx.x >> 5;
#pragma unroll
    for (int i = 0; i < 4; ++i) {
        const int k = k0 + ty * 4 + i;
        const float* src = (k < D_) ? (Ul + (size_t)k * NW_)
                                    : (Ur + (size_t)(k - D_) * NW_);
        tile[ty * 4 + i][tx] = src[n0 + tx];
    }
    __syncthreads();
#pragma unroll
    for (int i = 0; i < 4; ++i) {
        const int n = n0 + ty * 4 + i;          // gate-major column id
        const int g = n >> 10;
        const int d = n & 1023;
        const int r = (d >> 4) * ROWS_ + g * 16 + (d & 15);
        Wp[(size_t)r * K_ + k0 + tx] = (__bf16)tile[tx][ty * 4 + i];
    }
}

// ---------------------------------------------------------------------------
// Kernel 2: initial recurrent state = last leaf (seq[:,63,:]).
// ---------------------------------------------------------------------------
__global__ __launch_bounds__(256) void init_state(const float* __restrict__ seq,
                                                  float* __restrict__ h0,
                                                  float* __restrict__ c0) {
    const int idx = blockIdx.x * 256 + threadIdx.x;
    const int m = idx >> 10;
    const int d = idx & 1023;
    const size_t base = ((size_t)m * L_ + (L_ - 1)) * (2 * D_);
    h0[idx] = seq[base + d];
    c0[idx] = seq[base + D_ + d];
}

// ---------------------------------------------------------------------------
// Kernel 3: one reduce step. Block = 8 waves, all sharing d-tile dt=blockIdx,
// wave w owns m-tile w. TDM double-buffers the shared 80x128 B panel into LDS;
// waves ds_load B fragments, build A fragments from global (leaf / recurrent
// h), accumulate 5 gate tiles with v_wmma bf16, then apply the TreeLSTM cell.
// ---------------------------------------------------------------------------
__global__ __launch_bounds__(256) void fused_step(const float* __restrict__ seq,
                                                  const float* __restrict__ bias,
                                                  const __bf16* __restrict__ Wp,
                                                  const float* __restrict__ h_in,
                                                  const float* __restrict__ c_in,
                                                  float* __restrict__ h_out,
                                                  float* __restrict__ c_out,
                                                  int t) {
    const int tid  = threadIdx.x;
    const int lane = tid & 31;
    const int wave = tid >> 5;
    const int mt   = wave;                    // 0..7
    const int dt   = blockIdx.x;              // 0..63
    const int lrow = lane & 15;
    const int hi   = lane >> 4;               // lane half
    const int m    = mt * 16 + lrow;          // A row for this lane
    const int dcol = dt * 16 + lrow;          // output column for this lane
    const int leaf = 62 - t;

    const float* aLeaf = seq + ((size_t)m * L_ + leaf) * (2 * D_);
    const float* aRec  = h_in + (size_t)m * D_;

    const v8f zero = {0.f, 0.f, 0.f, 0.f, 0.f, 0.f, 0.f, 0.f};
    v8f acc[5];
#pragma unroll
    for (int g = 0; g < 5; ++g) acc[g] = zero;

    const int kb = hi * 16;  // A K sub-offset (lanes 0-15: K0-15, 16-31: K16-31)

#if HAVE_TDM
    __shared__ __align__(256) unsigned char sB[2][BUFB_];
    const __bf16* panel = Wp + (size_t)dt * ROWS_ * K_;
    const unsigned lds0 = (unsigned)(unsigned long long)&sB[0][0];
    const unsigned lds1 = (unsigned)(unsigned long long)&sB[1][0];

    int buf = 0;
    if (wave == 0) tdm_load_tile(panel, lds0);
    for (int c = 0; c < NCHUNK; ++c) {
        if (wave == 0) {
            if (c + 1 < NCHUNK) {
                tdm_load_tile(panel + (c + 1) * KC_, (buf ^ 1) ? lds1 : lds0);
                WAIT_TENSOR(1);     // chunk c complete (TDM is in-order)
            } else {
                WAIT_TENSOR(0);
            }
        }
        __syncthreads();            // chunk c visible in sB[buf]

        const float* srcBase = (c < NCHUNK / 2) ? (aLeaf + c * KC_)
                                                : (aRec + (c - NCHUNK / 2) * KC_);
        const unsigned char* bbase = sB[buf];
#pragma unroll
        for (int ks = 0; ks < KC_ / 32; ++ks) {
            const float* src = srcBase + ks * 32;
            v16bf afrag;
#pragma unroll
            for (int e = 0; e < 8; ++e) afrag[e]     = (__bf16)src[kb + e];
#pragma unroll
            for (int e = 0; e < 8; ++e) afrag[8 + e] = (__bf16)src[kb + 8 + e];
#pragma unroll
            for (int g = 0; g < 5; ++g) {
                const unsigned char* bp =
                    bbase + (g * 16 + lrow) * RSTRIDE + ks * 64 + hi * 32;
                const v8bf blo = *(const v8bf*)(bp);
                const v8bf bhi = *(const v8bf*)(bp + 16);
                const v16bf bfrag = __builtin_shufflevector(
                    blo, bhi, 0, 1, 2, 3, 4, 5, 6, 7, 8, 9, 10, 11, 12, 13, 14, 15);
                acc[g] = __builtin_amdgcn_wmma_f32_16x16x32_bf16(
                    false, afrag, false, bfrag, (short)0, acc[g], false, false);
            }
        }
        __syncthreads();            // all reads of sB[buf] done before reuse
        buf ^= 1;
    }
#else
    // Fallback: direct global B fragments (same packed layout, served by L2).
    for (int kk = 0; kk < K_; kk += 32) {
        const float* src = (kk < D_) ? (aLeaf + kk) : (aRec + (kk - D_));
        v16bf afrag;
#pragma unroll
        for (int e = 0; e < 8; ++e) afrag[e]     = (__bf16)src[kb + e];
#pragma unroll
        for (int e = 0; e < 8; ++e) afrag[8 + e] = (__bf16)src[kb + 8 + e];
#pragma unroll
        for (int g = 0; g < 5; ++g) {
            const __bf16* bp = Wp + (size_t)(dt * ROWS_ + g * 16 + lrow) * K_
                             + kk + hi * 16;
            const v16bf bfrag = *(const v16bf*)bp;
            acc[g] = __builtin_amdgcn_wmma_f32_16x16x32_bf16(
                false, afrag, false, bfrag, (short)0, acc[g], false, false);
        }
    }
#endif

    // Fused TreeLSTM cell epilogue. C/D layout: VGPR j + lane half -> row.
    const float bi  = bias[dcol];
    const float bo  = bias[D_ + dcol];
    const float bfl = bias[2 * D_ + dcol];
    const float bfr = bias[3 * D_ + dcol];
    const float bu  = bias[4 * D_ + dcol];
#pragma unroll
    for (int j = 0; j < 8; ++j) {
        const int mm = mt * 16 + j + hi * 8;
        const float ig = acc[0][j] + bi;
        const float og = acc[1][j] + bo;
        const float fl = acc[2][j] + bfl;
        const float fr = acc[3][j] + bfr;
        const float ug = acc[4][j] + bu;
        const float cl = seq[((size_t)mm * L_ + leaf) * (2 * D_) + D_ + dcol];
        const float cr = c_in[(size_t)mm * D_ + dcol];
        const float cj = sigmoidf(ig) * tanhf(ug) + sigmoidf(fl) * cl
                       + sigmoidf(fr) * cr;
        const float hj = sigmoidf(og) * tanhf(cj);
        h_out[(size_t)mm * D_ + dcol] = hj;
        c_out[(size_t)mm * D_ + dcol] = cj;
    }
}

// ---------------------------------------------------------------------------
extern "C" void kernel_launch(void* const* d_in, const int* in_sizes, int n_in,
                              void* d_out, int out_size, void* d_ws, size_t ws_size,
                              hipStream_t stream) {
    const float* seq = (const float*)d_in[0];
    // d_in[1] = transitions (int32): static 64-shift/63-reduce schedule.
    const float* Ulw = (const float*)d_in[2];
    const float* Ulb = (const float*)d_in[3];
    const float* Urw = (const float*)d_in[4];

    char* ws = (char*)d_ws;
    __bf16* Wp = (__bf16*)ws;                          // 5120*2048*2 = 20 MB
    size_t off = (size_t)NW_ * K_ * sizeof(__bf16);
    float* h0 = (float*)(ws + off); off += (size_t)B_ * D_ * sizeof(float);
    float* h1 = (float*)(ws + off); off += (size_t)B_ * D_ * sizeof(float);
    float* c0 = (float*)(ws + off); off += (size_t)B_ * D_ * sizeof(float);
    float* c1 = (float*)(ws + off); off += (size_t)B_ * D_ * sizeof(float);

    dim3 gW(K_ / 32, NW_ / 32);
    prep_weights<<<gW, 256, 0, stream>>>(Ulw, Urw, Wp);
    init_state<<<(B_ * D_) / 256, 256, 0, stream>>>(seq, h0, c0);

    float* hbuf[2] = {h0, h1};
    float* cbuf[2] = {c0, c1};
    for (int t = 0; t < NSTEP; ++t) {
        float* hout = (t == NSTEP - 1) ? (float*)d_out : hbuf[(t + 1) & 1];
        fused_step<<<64, 256, 0, stream>>>(seq, Ulb, Wp,
                                           hbuf[t & 1], cbuf[t & 1],
                                           hout, cbuf[(t + 1) & 1], t);
    }
}
